// GraphTransformerModule_12644383719678
// MI455X (gfx1250) — matrix operand order repaired and checked by reading, not compiled, not run on hardware
//
#include <hip/hip_runtime.h>

#define NN 50000
#define EE 800000
#define CC 128
#define HH 8

typedef __attribute__((ext_vector_type(16))) __bf16 v16bf;
typedef __attribute__((ext_vector_type(8)))  float  v8f;

// ---------- helpers ----------
__device__ __forceinline__ int rfl(int v) { return __builtin_amdgcn_readfirstlane(v); }

__device__ __forceinline__ unsigned short f2bfu(float f) {
    unsigned int u = __float_as_uint(f);
    unsigned int r = u + 0x7FFFu + ((u >> 16) & 1u);   // round-to-nearest-even
    return (unsigned short)(r >> 16);
}
__device__ __forceinline__ unsigned int pack_bf2(float lo, float hi) {
    return (unsigned int)f2bfu(lo) | ((unsigned int)f2bfu(hi) << 16);
}
__device__ __forceinline__ float clip5(float x) { return fminf(fmaxf(x, -5.0f), 5.0f); }
__device__ __forceinline__ float siluf(float x) { return x / (1.0f + __expf(-x)); }

// A-fragment: 16xK bf16 tile in LDS, row-major (rowStride in ushorts).
// ISA 7.12.2 16-bit A 16x32: lane m=lane&15, half=lane>>4; VGPR j: g=j>>2,p=j&3,
// K = kc*32 + g*16 + half*8 + p*2  (pair)
__device__ __forceinline__ v16bf load_a_frag(const unsigned short* Abase, int lane,
                                             int kc, int rowStride) {
    int m = lane & 15, half = lane >> 4;
    union { v16bf v; unsigned int u[8]; } a;
#pragma unroll
    for (int j = 0; j < 8; ++j) {
        int g = j >> 2, p = j & 3;
        int k0 = kc * 32 + g * 16 + half * 8 + p * 2;
        a.u[j] = *(const unsigned int*)(Abase + m * rowStride + k0);
    }
    return a.v;
}

// B-fragment from pre-packed weights: layout [nt][kc][lane][j] dwords -> lane reads
// 32 contiguous bytes (2x b128, fully coalesced).
__device__ __forceinline__ v16bf load_b_frag(const unsigned int* Wpk, int lane,
                                             int nt, int kc, int KC) {
    const uint4* p4 = (const uint4*)(Wpk + (((nt * KC) + kc) * 32 + lane) * 8);
    uint4 x0 = p4[0], x1 = p4[1];
    union { v16bf v; unsigned int u[8]; } b;
    b.u[0]=x0.x; b.u[1]=x0.y; b.u[2]=x0.z; b.u[3]=x0.w;
    b.u[4]=x1.x; b.u[5]=x1.y; b.u[6]=x1.z; b.u[7]=x1.w;
    return b.v;
}

template <int KC>
__device__ __forceinline__ v8f gemm_tile(const unsigned short* Abase, int rowStride,
                                         const unsigned int* Wpk, int nt, int lane) {
    v8f acc = {};
#pragma unroll
    for (int g0 = 0; g0 < KC; g0 += 4) {
        v16bf b[4], a[4];
#pragma unroll
        for (int kc = 0; kc < 4; ++kc) b[kc] = load_b_frag(Wpk, lane, nt, g0 + kc, KC);
#pragma unroll
        for (int kc = 0; kc < 4; ++kc) a[kc] = load_a_frag(Abase, lane, g0 + kc, rowStride);
#pragma unroll
        for (int kc = 0; kc < 4; ++kc)
            acc = __builtin_amdgcn_wmma_f32_16x16x32_bf16(false, a[kc], false, b[kc],
                                                          (short)0, acc, false, false);
    }
    return acc;
}

// LayerNorm one 128-wide row (lane owns 4 channels), write bf16 pairs to LDS row.
__device__ __forceinline__ void ln128_row(const float* src, const float* g,
                                          const float* b, unsigned int* dstPairs,
                                          int lane) {
    const float4 xv = *(const float4*)(src + lane * 4);
    float s = xv.x + xv.y + xv.z + xv.w;
#pragma unroll
    for (int m = 16; m >= 1; m >>= 1) s += __shfl_xor(s, m, 32);
    float mu = s * (1.0f / 128.0f);
    float d0 = xv.x - mu, d1 = xv.y - mu, d2 = xv.z - mu, d3 = xv.w - mu;
    float vs = d0*d0 + d1*d1 + d2*d2 + d3*d3;
#pragma unroll
    for (int m = 16; m >= 1; m >>= 1) vs += __shfl_xor(vs, m, 32);
    float inv = rsqrtf(vs * (1.0f / 128.0f) + 1e-5f);
    const float4 gv = *(const float4*)(g + lane * 4);
    const float4 bv = *(const float4*)(b + lane * 4);
    dstPairs[lane * 2 + 0] = pack_bf2(d0 * inv * gv.x + bv.x, d1 * inv * gv.y + bv.y);
    dstPairs[lane * 2 + 1] = pack_bf2(d2 * inv * gv.z + bv.z, d3 * inv * gv.w + bv.w);
}

// ---------- prep kernels ----------
__global__ void zero_kernel(float* p, int n) {
    int i = blockIdx.x * blockDim.x + threadIdx.x;
    if (i < n) p[i] = 0.0f;
}

// Repack fp32 W[K][Nc] (row-major) into B-fragment order [nt][kc][lane][j] bf16 pairs.
__global__ void repack_weight(const float* __restrict__ W, unsigned int* __restrict__ out,
                              int K, int Nc) {
    int idx = blockIdx.x * blockDim.x + threadIdx.x;
    int total = (K * Nc) >> 1;
    if (idx >= total) return;
    int KC   = K >> 5;
    int j    = idx & 7;
    int lane = (idx >> 3) & 31;
    int rest = idx >> 8;
    int kc   = rest % KC;
    int nt   = rest / KC;
    int n    = nt * 16 + (lane & 15);
    int half = lane >> 4;
    int k    = kc * 32 + (j >> 2) * 16 + half * 8 + (j & 3) * 2;
    out[idx] = pack_bf2(W[(long long)k * Nc + n], W[(long long)(k + 1) * Nc + n]);
}

// ---------- node Q/K/V: LN1 + 3 GEMMs ----------
__global__ __launch_bounds__(64) void node_qkv_kernel(
        const float* __restrict__ x, const float* __restrict__ g, const float* __restrict__ b,
        const unsigned int* WqP, const unsigned int* WkP, const unsigned int* WvP,
        float* __restrict__ qb, float* __restrict__ kb, float* __restrict__ vb) {
    __shared__ __align__(16) unsigned short As_s[2][16 * 128];
    int lane = threadIdx.x & 31;
    int wave = rfl(threadIdx.x >> 5);                 // scalar (SGPR) wave id
    int n0   = rfl((blockIdx.x * 2 + wave) * 16);     // scalar tile base
    bool full = (n0 + 16 <= NN);                      // scalar branch
    unsigned short* As = As_s[wave];
    __builtin_prefetch(WqP, 0, 1);

    for (int r = 0; r < 16; ++r) {
        int node = n0 + r; if (node >= NN) node = NN - 1;
        ln128_row(x + (long long)node * CC, g, b, (unsigned int*)As + r * 64, lane);
    }
    __syncthreads();

    const unsigned int* Ws[3] = {WqP, WkP, WvP};
    float* Ds[3] = {qb, kb, vb};
#pragma unroll
    for (int w = 0; w < 3; ++w) {
        for (int nt = 0; nt < 8; ++nt) {
            v8f acc = gemm_tile<4>(As, 128, Ws[w], nt, lane);
            int n = nt * 16 + (lane & 15), half = lane >> 4;
            if (full) {
#pragma unroll
                for (int r2 = 0; r2 < 8; ++r2)
                    Ds[w][(long long)(n0 + r2 + half * 8) * CC + n] = acc[r2];
            } else {
#pragma unroll
                for (int r2 = 0; r2 < 8; ++r2) {
                    int node = n0 + r2 + half * 8;
                    if (node < NN) Ds[w][(long long)node * CC + n] = acc[r2];
                }
            }
        }
    }
}

// ---------- fused edge kernel ----------
__global__ __launch_bounds__(64) void edge_kernel(
        const float* __restrict__ edge_attr, const int* __restrict__ edge_index,
        const float* ln1g, const float* ln1b,
        const unsigned int* WeP, const unsigned int* WoeP, const float* __restrict__ bo_e,
        const float* ln2g, const float* ln2b,
        const unsigned int* W1eP, const unsigned int* W2eP,
        const float* __restrict__ qb, const float* __restrict__ kb, const float* __restrict__ vb,
        float* __restrict__ wV, float* __restrict__ zb, float* __restrict__ out_e) {
    __shared__ __align__(16) unsigned short As_s[2][16 * 128];   // LN1/LN2 bf16 tile
    __shared__ __align__(16) float          EpU_s[2][16 * 128];  // ep fp32; later U bf16[16][256]
    __shared__ __align__(16) unsigned short Eo_s[2][16 * 128];   // e_out bf16 tile
    __shared__ __align__(16) float          Er_s[2][16 * 128];   // residual e fp32
    int lane = threadIdx.x & 31;
    int wave = rfl(threadIdx.x >> 5);                 // scalar
    int e0   = rfl((blockIdx.x * 2 + wave) * 16);     // scalar tile base
    unsigned short* As = As_s[wave];
    float* Ep = EpU_s[wave];
    unsigned short* Us = (unsigned short*)EpU_s[wave];
    unsigned short* Eo = Eo_s[wave];
    float* Er = Er_s[wave];
    __builtin_prefetch(WeP, 0, 1);

    // LN1(edge_attr) -> As (bf16)
    for (int r = 0; r < 16; ++r)
        ln128_row(edge_attr + (long long)(e0 + r) * CC, ln1g, ln1b,
                  (unsigned int*)As + r * 64, lane);
    __syncthreads();

    // ep = LN(e) @ We  -> Ep (fp32 LDS)
    for (int nt = 0; nt < 8; ++nt) {
        v8f acc = gemm_tile<4>(As, 128, WeP, nt, lane);
        int n = nt * 16 + (lane & 15), half = lane >> 4;
#pragma unroll
        for (int r2 = 0; r2 < 8; ++r2) Ep[(r2 + half * 8) * CC + n] = acc[r2];
    }
    __syncthreads();

    // attention elementwise + scatter-add (q/k/v are L2-resident)
    for (int r = 0; r < 16; ++r) {
        int e   = e0 + r;
        int src = edge_index[e];
        int dst = edge_index[EE + e];
        float4 kk = *(const float4*)(kb + (long long)src * CC + lane * 4);
        float4 qq = *(const float4*)(qb + (long long)dst * CC + lane * 4);
        float4 ep = *(const float4*)(Ep + r * CC + lane * 4);
        float a0 = clip5(kk.x * qq.x * 0.25f) * ep.x;
        float a1 = clip5(kk.y * qq.y * 0.25f) * ep.y;
        float a2 = clip5(kk.z * qq.z * 0.25f) * ep.z;
        float a3 = clip5(kk.w * qq.w * 0.25f) * ep.w;
        unsigned int* eo = (unsigned int*)Eo + r * 64;
        eo[lane * 2 + 0] = pack_bf2(a0, a1);
        eo[lane * 2 + 1] = pack_bf2(a2, a3);
        float hs = a0 + a1 + a2 + a3;                       // per-head (D=16) sum
        hs += __shfl_xor(hs, 1, 32);
        hs += __shfl_xor(hs, 2, 32);
        float ax = __expf(clip5(hs));
        float4 vv = *(const float4*)(vb + (long long)src * CC + lane * 4);
        float* wd = wV + (long long)dst * CC + lane * 4;
        atomicAdd(wd + 0, vv.x * ax);
        atomicAdd(wd + 1, vv.y * ax);
        atomicAdd(wd + 2, vv.z * ax);
        atomicAdd(wd + 3, vv.w * ax);
        if ((lane & 3) == 0) atomicAdd(zb + (long long)dst * HH + (lane >> 2), ax);
    }
    __syncthreads();

    // e_res = e_out @ Wo_e + bo_e + edge_attr  -> Er
    for (int nt = 0; nt < 8; ++nt) {
        v8f acc = gemm_tile<4>(Eo, 128, WoeP, nt, lane);
        int n = nt * 16 + (lane & 15), half = lane >> 4;
#pragma unroll
        for (int r2 = 0; r2 < 8; ++r2) {
            int m = r2 + half * 8;
            Er[m * CC + n] = acc[r2] + bo_e[n] + edge_attr[(long long)(e0 + m) * CC + n];
        }
    }
    __syncthreads();

    // LN2 -> As
    for (int r = 0; r < 16; ++r)
        ln128_row(Er + r * CC, ln2g, ln2b, (unsigned int*)As + r * 64, lane);
    __syncthreads();

    // FFN: U = silu(As @ W1e)  (K=128 -> Nc=256)
    for (int nt = 0; nt < 16; ++nt) {
        v8f acc = gemm_tile<4>(As, 128, W1eP, nt, lane);
        int n = nt * 16 + (lane & 15), half = lane >> 4;
#pragma unroll
        for (int r2 = 0; r2 < 8; ++r2)
            Us[(r2 + half * 8) * 256 + n] = f2bfu(siluf(acc[r2]));
    }
    __syncthreads();

    // out = Er + U @ W2e   (K=256); non-temporal: never re-read, keep L2 for q/k/v
    for (int nt = 0; nt < 8; ++nt) {
        v8f acc = gemm_tile<8>(Us, 256, W2eP, nt, lane);
        int n = nt * 16 + (lane & 15), half = lane >> 4;
#pragma unroll
        for (int r2 = 0; r2 < 8; ++r2) {
            int m = r2 + half * 8;
            __builtin_nontemporal_store(Er[m * CC + n] + acc[r2],
                                        &out_e[(long long)(e0 + m) * CC + n]);
        }
    }
}

// ---------- node epilogue: wV/z -> Wo_n -> LN2 -> FFN ----------
__global__ __launch_bounds__(64) void node_out_kernel(
        const float* __restrict__ x, const float* __restrict__ wV, const float* __restrict__ zb,
        const unsigned int* WonP, const float* __restrict__ bo_n,
        const float* ln2g, const float* ln2b,
        const unsigned int* W1nP, const unsigned int* W2nP, float* __restrict__ out_h) {
    __shared__ __align__(16) unsigned short As_s[2][16 * 128];
    __shared__ __align__(16) float          Hr_s[2][16 * 128];
    __shared__ __align__(16) unsigned short U_s [2][16 * 256];
    int lane = threadIdx.x & 31;
    int wave = rfl(threadIdx.x >> 5);                 // scalar
    int n0   = rfl((blockIdx.x * 2 + wave) * 16);     // scalar
    bool full = (n0 + 16 <= NN);                      // scalar branch
    unsigned short* As = As_s[wave];
    float* Hr = Hr_s[wave];
    unsigned short* Us = U_s[wave];
    __builtin_prefetch(WonP, 0, 1);

    // h_attn = wV / (z + 1e-6) -> bf16 A-tile
    for (int r = 0; r < 16; ++r) {
        int node = n0 + r; if (node >= NN) node = NN - 1;
        float4 wv = *(const float4*)(wV + (long long)node * CC + lane * 4);
        float inv = 1.0f / (zb[(long long)node * HH + (lane >> 2)] + 1e-6f);
        unsigned int* dp = (unsigned int*)As + r * 64;
        dp[lane * 2 + 0] = pack_bf2(wv.x * inv, wv.y * inv);
        dp[lane * 2 + 1] = pack_bf2(wv.z * inv, wv.w * inv);
    }
    __syncthreads();

    // h_res = h_attn @ Wo_n + bo_n + x
    for (int nt = 0; nt < 8; ++nt) {
        v8f acc = gemm_tile<4>(As, 128, WonP, nt, lane);
        int n = nt * 16 + (lane & 15), half = lane >> 4;
#pragma unroll
        for (int r2 = 0; r2 < 8; ++r2) {
            int m = r2 + half * 8;
            int node = n0 + m; if (node >= NN) node = NN - 1;
            Hr[m * CC + n] = acc[r2] + bo_n[n] + x[(long long)node * CC + n];
        }
    }
    __syncthreads();

    for (int r = 0; r < 16; ++r)
        ln128_row(Hr + r * CC, ln2g, ln2b, (unsigned int*)As + r * 64, lane);
    __syncthreads();

    for (int nt = 0; nt < 16; ++nt) {
        v8f acc = gemm_tile<4>(As, 128, W1nP, nt, lane);
        int n = nt * 16 + (lane & 15), half = lane >> 4;
#pragma unroll
        for (int r2 = 0; r2 < 8; ++r2)
            Us[(r2 + half * 8) * 256 + n] = f2bfu(siluf(acc[r2]));
    }
    __syncthreads();

    for (int nt = 0; nt < 8; ++nt) {
        v8f acc = gemm_tile<8>(Us, 256, W2nP, nt, lane);
        int n = nt * 16 + (lane & 15), half = lane >> 4;
        if (full) {
#pragma unroll
            for (int r2 = 0; r2 < 8; ++r2) {
                int m = r2 + half * 8;
                __builtin_nontemporal_store(Hr[m * CC + n] + acc[r2],
                                            &out_h[(long long)(n0 + m) * CC + n]);
            }
        } else {
#pragma unroll
            for (int r2 = 0; r2 < 8; ++r2) {
                int m = r2 + half * 8, node = n0 + m;
                if (node < NN) out_h[(long long)node * CC + n] = Hr[m * CC + n] + acc[r2];
            }
        }
    }
}

// ---------- host ----------
extern "C" void kernel_launch(void* const* d_in, const int* in_sizes, int n_in,
                              void* d_out, int out_size, void* d_ws, size_t ws_size,
                              hipStream_t stream) {
    const float* x         = (const float*)d_in[0];
    const float* edge_attr = (const float*)d_in[1];
    const int*   edge_index= (const int*)  d_in[2];
    const float* Wq  = (const float*)d_in[3];
    const float* Wk  = (const float*)d_in[4];
    const float* Wv  = (const float*)d_in[5];
    const float* We  = (const float*)d_in[6];
    const float* ln1n_g = (const float*)d_in[7];
    const float* ln1n_b = (const float*)d_in[8];
    const float* ln1e_g = (const float*)d_in[9];
    const float* ln1e_b = (const float*)d_in[10];
    const float* Wo_n = (const float*)d_in[11];
    const float* bo_n = (const float*)d_in[12];
    const float* Wo_e = (const float*)d_in[13];
    const float* bo_e = (const float*)d_in[14];
    const float* ln2n_g = (const float*)d_in[15];
    const float* ln2n_b = (const float*)d_in[16];
    const float* ln2e_g = (const float*)d_in[17];
    const float* ln2e_b = (const float*)d_in[18];
    const float* W1n = (const float*)d_in[19];
    const float* W2n = (const float*)d_in[20];
    const float* W1e = (const float*)d_in[21];
    const float* W2e = (const float*)d_in[22];

    float* out_h = (float*)d_out;
    float* out_e = out_h + (size_t)NN * CC;

    // workspace layout (dwords)
    unsigned int* u = (unsigned int*)d_ws;
    unsigned int* WqP  = u + 0;
    unsigned int* WkP  = u + 8192;
    unsigned int* WvP  = u + 16384;
    unsigned int* WeP  = u + 24576;
    unsigned int* WonP = u + 32768;
    unsigned int* WoeP = u + 40960;
    unsigned int* W1nP = u + 49152;
    unsigned int* W2nP = u + 65536;
    unsigned int* W1eP = u + 81920;
    unsigned int* W2eP = u + 98304;
    float* f = (float*)d_ws;
    float* qb  = f + 114688;
    float* kb  = qb + (size_t)NN * CC;
    float* vb  = kb + (size_t)NN * CC;
    float* wV  = vb + (size_t)NN * CC;
    float* zbuf= wV + (size_t)NN * CC;

    // repack weights to bf16 B-fragment layout (L2-resident, ~450 KB total)
    repack_weight<<<32, 256, 0, stream>>>(Wq,   WqP, 128, 128);
    repack_weight<<<32, 256, 0, stream>>>(Wk,   WkP, 128, 128);
    repack_weight<<<32, 256, 0, stream>>>(Wv,   WvP, 128, 128);
    repack_weight<<<32, 256, 0, stream>>>(We,   WeP, 128, 128);
    repack_weight<<<32, 256, 0, stream>>>(Wo_n, WonP, 128, 128);
    repack_weight<<<32, 256, 0, stream>>>(Wo_e, WoeP, 128, 128);
    repack_weight<<<64, 256, 0, stream>>>(W1n,  W1nP, 128, 256);
    repack_weight<<<64, 256, 0, stream>>>(W2n,  W2nP, 256, 128);
    repack_weight<<<64, 256, 0, stream>>>(W1e,  W1eP, 128, 256);
    repack_weight<<<64, 256, 0, stream>>>(W2e,  W2eP, 256, 128);

    // zero accumulators (every call: harness doesn't reset workspace)
    int zn = NN * CC + NN * HH;
    zero_kernel<<<(zn + 255) / 256, 256, 0, stream>>>(wV, zn);

    node_qkv_kernel<<<(NN + 31) / 32, 64, 0, stream>>>(
        x, ln1n_g, ln1n_b, WqP, WkP, WvP, qb, kb, vb);

    edge_kernel<<<EE / 32, 64, 0, stream>>>(
        edge_attr, edge_index, ln1e_g, ln1e_b, WeP, WoeP, bo_e,
        ln2e_g, ln2e_b, W1eP, W2eP, qb, kb, vb, wV, zbuf, out_e);

    node_out_kernel<<<(NN + 31) / 32, 64, 0, stream>>>(
        x, wV, zbuf, WonP, bo_n, ln2n_g, ln2n_b, W1nP, W2nP, out_h);
}
